// Head_47390669144790
// MI455X (gfx1250) — compile-verified, matrix operand order checked by reading, not run on hardware
//
#include <hip/hip_runtime.h>
#include <hip/hip_bf16.h>

// CDNA5 (gfx1250) wave32 WMMA fragments
typedef _Float16 v16h __attribute__((ext_vector_type(16)));
typedef _Float16 v8h  __attribute__((ext_vector_type(8)));
typedef _Float16 v4h  __attribute__((ext_vector_type(4)));
typedef float    v8f  __attribute__((ext_vector_type(8)));

#define B_ 4
#define T_ 4096
#define C_ 1024
#define H_ 64

__device__ __forceinline__ v16h cat8(v8h lo, v8h hi) {
    return __builtin_shufflevector(lo, hi, 0, 1, 2, 3, 4, 5, 6, 7,
                                          8, 9, 10, 11, 12, 13, 14, 15);
}

__device__ __forceinline__ v8f wmma_f16(v16h a, v16h b, v8f c) {
    // (neg_a, A, neg_b, B, c_mod, C, reuse_a, reuse_b)
    return __builtin_amdgcn_wmma_f32_16x16x32_f16(false, a, false, b,
                                                  (short)0, c, false, false);
}

// ---------------------------------------------------------------------------
// Kernel 1: Q/K/V projections.  x[16384,1024](f32) @ W[1024,64](f32) -> f16.
// Block = 128 threads (4 waves); each block does 64 rows, each wave 16 rows.
// The p-loop is fully unrolled so W/out pointers stay in the global address
// space (GLOBAL_* instead of FLAT_*, keeping DScnt free for fragment loads).
// Softmax scale 1/sqrt(64)=0.125 folded into the Q store (exact pow2).
// ---------------------------------------------------------------------------
__global__ __launch_bounds__(128) void qkv_proj_wmma(
    const float* __restrict__ x,
    const float* __restrict__ Wq, const float* __restrict__ Wk,
    const float* __restrict__ Wv,
    _Float16* __restrict__ Qh, _Float16* __restrict__ Kh,
    _Float16* __restrict__ Vh)
{
    __shared__ __align__(16) _Float16 xs[64][32];   // x chunk  (row-major, f16)
    __shared__ __align__(16) _Float16 wt[64][32];   // W chunk  transposed [h][k]

    const int tid  = threadIdx.x;
    const int lane = tid & 31;
    const int wv   = tid >> 5;          // wave id 0..3
    const int rowBase = blockIdx.x * 64;
    const int m  = lane & 15;           // row (A) / col (B) within fragment
    const int hf = lane >> 4;           // lane half selects K sub-range

    #pragma unroll
    for (int p = 0; p < 3; ++p) {
        const float* W = (p == 0) ? Wq : (p == 1) ? Wk : Wv;
        _Float16* outp = (p == 0) ? Qh : (p == 1) ? Kh : Vh;

        v8f acc[4];
        #pragma unroll
        for (int j = 0; j < 4; ++j)
            for (int r = 0; r < 8; ++r) acc[j][r] = 0.0f;

        for (int c0 = 0; c0 < C_; c0 += 32) {
            // stage 64x32 x-chunk as f16: 512 float4 segments, 4 per thread
            #pragma unroll
            for (int s = 0; s < 4; ++s) {
                int idx = tid + s * 128;          // 0..511
                int rr  = idx >> 3;               // 0..63
                int cc  = (idx & 7) * 4;          // 0,4,..,28
                const float4 f =
                    *(const float4*)&x[(size_t)(rowBase + rr) * C_ + c0 + cc];
                v4h h;
                h[0] = (_Float16)f.x; h[1] = (_Float16)f.y;
                h[2] = (_Float16)f.z; h[3] = (_Float16)f.w;
                *(v4h*)&xs[rr][cc] = h;
            }
            // stage 32x64 W-chunk transposed (wt[h][k]); coalesced float4 reads
            #pragma unroll
            for (int s = 0; s < 4; ++s) {
                int idx = tid + s * 128;          // 0..511
                int kk  = idx >> 4;               // 0..31
                int hh  = (idx & 15) * 4;         // 0,4,..,60
                const float4 f =
                    *(const float4*)&W[(size_t)(c0 + kk) * H_ + hh];
                wt[hh + 0][kk] = (_Float16)f.x;
                wt[hh + 1][kk] = (_Float16)f.y;
                wt[hh + 2][kk] = (_Float16)f.z;
                wt[hh + 3][kk] = (_Float16)f.w;
            }
            __syncthreads();

            // A fragment: lane m holds row m, K-elements per ISA 16-bit layout
            const v8h* arow = (const v8h*)&xs[wv * 16 + m][0];
            v16h a = cat8(arow[hf], arow[2 + hf]);
            #pragma unroll
            for (int j = 0; j < 4; ++j) {
                const v8h* brow = (const v8h*)&wt[j * 16 + m][0];
                v16h b = cat8(brow[hf], brow[2 + hf]);
                acc[j] = wmma_f16(a, b, acc[j]);
            }
            __syncthreads();
        }

        // C/D layout: lane L vgpr r -> row (r + 8*(L>>4)), col (L&15)
        const float sc = (p == 0) ? 0.125f : 1.0f;   // fold 1/sqrt(H) into Q
        #pragma unroll
        for (int j = 0; j < 4; ++j)
            #pragma unroll
            for (int r = 0; r < 8; ++r) {
                int q = rowBase + wv * 16 + r + 8 * hf;
                outp[(size_t)q * H_ + j * 16 + m] = (_Float16)(acc[j][r] * sc);
            }
        __syncthreads();
    }
}

// ---------------------------------------------------------------------------
// Kernel 2: causal flash attention.  Block = 64 queries (4 waves x 16),
// key tiles of 64 staged in LDS (K via async global->LDS b128, V transposed
// through registers); online softmax; O accumulated with WMMA.
// ---------------------------------------------------------------------------
__global__ __launch_bounds__(128) void flash_attn_wmma(
    const _Float16* __restrict__ Qh, const _Float16* __restrict__ Kh,
    const _Float16* __restrict__ Vh, float* __restrict__ out)
{
    __shared__ __align__(16) _Float16 Ks[64][64];     // K tile [key][h]   8 KB
    __shared__ __align__(16) _Float16 Vt[64][64];     // V tile [h][key]   8 KB
    __shared__ __align__(16) _Float16 Ps[4][16][64];  // per-wave P scratch 8 KB

    const int tid  = threadIdx.x;
    const int lane = tid & 31;
    const int wv   = tid >> 5;
    const int m    = lane & 15;
    const int hf   = lane >> 4;
    const int b    = blockIdx.y;
    const int qt   = blockIdx.x;          // query tile within batch (0..63)
    const int q0   = qt * 64 + wv * 16;   // this wave's first query row
    const size_t base = (size_t)b * T_ * H_;

    // Q fragments (A operand), contraction over H=64 -> two K=32 chunks.
    // Q already carries the 1/sqrt(H) scale from the projection kernel.
    v16h qa0, qa1;
    {
        const v8h* qrow = (const v8h*)(Qh + base + (size_t)(q0 + m) * H_);
        qa0 = cat8(qrow[hf],     qrow[2 + hf]);
        qa1 = cat8(qrow[4 + hf], qrow[6 + hf]);
    }

    v8f oacc[4];
    #pragma unroll
    for (int j = 0; j < 4; ++j)
        for (int r = 0; r < 8; ++r) oacc[j][r] = 0.0f;
    float mrow[8], lrow[8];
    for (int r = 0; r < 8; ++r) { mrow[r] = -3.0e38f; lrow[r] = 0.0f; }

    const int ntiles = qt + 1;   // block-uniform causal frontier
    for (int kt = 0; kt < ntiles; ++kt) {
        const int kbase = kt * 64;

        // K tile: 512 x 16B segments, 4/thread, async global -> LDS (b128).
        #pragma unroll
        for (int s = 0; s < 4; ++s) {
            int idx = tid + s * 128;          // 0..511
            int kk  = idx >> 3;               // key row 0..63
            int h8  = (idx & 7) * 8;          // 8-half segment
            uint32_t lds   = (uint32_t)(uintptr_t)&Ks[kk][h8];
            uint64_t gaddr = (uint64_t)(uintptr_t)
                (Kh + base + (size_t)(kbase + kk) * H_ + h8);
            asm volatile("global_load_async_to_lds_b128 %0, %1, off"
                         :: "v"(lds), "v"(gaddr) : "memory");
        }
        // V tile: coalesced b128 reads, transposed 2B stores into Vt[h][key].
        #pragma unroll
        for (int s = 0; s < 4; ++s) {
            int idx = tid + s * 128;
            int kk  = idx >> 3;
            int h8  = (idx & 7) * 8;
            v8h v = *(const v8h*)(Vh + base + (size_t)(kbase + kk) * H_ + h8);
            #pragma unroll
            for (int j = 0; j < 8; ++j) Vt[h8 + j][kk] = v[j];
        }
        // drain ASYNCcnt (inline-asm ops are invisible to the compiler's
        // counter tracking), then block barrier.
        asm volatile("s_wait_asynccnt 0" ::: "memory");
        __syncthreads();

        // S = Q K^T for 4 key sub-tiles of 16 (B lane n = row n of K)
        v8f sacc[4];
        #pragma unroll
        for (int j = 0; j < 4; ++j) {
            for (int r = 0; r < 8; ++r) sacc[j][r] = 0.0f;
            const v8h* brow = (const v8h*)&Ks[j * 16 + m][0];
            v16h b0 = cat8(brow[hf],     brow[2 + hf]);
            v16h b1 = cat8(brow[4 + hf], brow[6 + hf]);
            sacc[j] = wmma_f16(qa0, b0, sacc[j]);
            sacc[j] = wmma_f16(qa1, b1, sacc[j]);
        }

        // causal mask (by value, EXEC stays full) + row max
        float rmax[8];
        for (int r = 0; r < 8; ++r) rmax[r] = -3.0e38f;
        #pragma unroll
        for (int j = 0; j < 4; ++j)
            #pragma unroll
            for (int r = 0; r < 8; ++r) {
                int kk = kbase + j * 16 + m;
                int qq = q0 + r + 8 * hf;
                float s = sacc[j][r];
                s = (kk <= qq) ? s : -3.0e38f;
                sacc[j][r] = s;
                rmax[r] = fmaxf(rmax[r], s);
            }
        // each score row lives in one 16-lane group -> width-16 xor reduction
        #pragma unroll
        for (int r = 0; r < 8; ++r)
            for (int off = 1; off < 16; off <<= 1)
                rmax[r] = fmaxf(rmax[r], __shfl_xor(rmax[r], off, 16));

        #pragma unroll
        for (int r = 0; r < 8; ++r) {
            float mn    = fmaxf(mrow[r], rmax[r]);
            float alpha = __expf(mrow[r] - mn);
            mrow[r] = mn;
            float psum = 0.0f;
            #pragma unroll
            for (int j = 0; j < 4; ++j) {
                float pv = __expf(sacc[j][r] - mn);
                sacc[j][r] = pv;
                psum += pv;
            }
            for (int off = 1; off < 16; off <<= 1)
                psum += __shfl_xor(psum, off, 16);
            lrow[r] = lrow[r] * alpha + psum;
            #pragma unroll
            for (int j = 0; j < 4; ++j) oacc[j][r] *= alpha;
        }

        // P (C/D layout) -> per-wave LDS, then reload as A fragments.
        #pragma unroll
        for (int j = 0; j < 4; ++j)
            #pragma unroll
            for (int r = 0; r < 8; ++r)
                Ps[wv][r + 8 * hf][j * 16 + m] = (_Float16)sacc[j][r];
        // wave-local DS ordering is enough (Ps is private to this wave)
        asm volatile("s_wait_dscnt 0" ::: "memory");

        v16h pa0, pa1;
        {
            const v8h* prow = (const v8h*)&Ps[wv][m][0];
            pa0 = cat8(prow[hf],     prow[2 + hf]);
            pa1 = cat8(prow[4 + hf], prow[6 + hf]);
        }
        // O += P @ V  (B lane n = column h of V = row of Vt)
        #pragma unroll
        for (int j = 0; j < 4; ++j) {
            const v8h* vrow = (const v8h*)&Vt[j * 16 + m][0];
            v16h vb0 = cat8(vrow[hf],     vrow[2 + hf]);
            v16h vb1 = cat8(vrow[4 + hf], vrow[6 + hf]);
            oacc[j] = wmma_f16(pa0, vb0, oacc[j]);
            oacc[j] = wmma_f16(pa1, vb1, oacc[j]);
        }
        __syncthreads();
    }

    // epilogue: normalize, store f32 non-temporally (written once, never read)
    #pragma unroll
    for (int j = 0; j < 4; ++j)
        #pragma unroll
        for (int r = 0; r < 8; ++r) {
            int qq = q0 + r + 8 * hf;
            __builtin_nontemporal_store(
                oacc[j][r] / lrow[r],
                &out[base + (size_t)qq * H_ + j * 16 + m]);
        }
}

// ---------------------------------------------------------------------------
extern "C" void kernel_launch(void* const* d_in, const int* in_sizes, int n_in,
                              void* d_out, int out_size, void* d_ws, size_t ws_size,
                              hipStream_t stream)
{
    const float* x  = (const float*)d_in[0];
    const float* Wq = (const float*)d_in[1];
    const float* Wk = (const float*)d_in[2];
    const float* Wv = (const float*)d_in[3];
    float* out = (float*)d_out;

    const size_t nQKV = (size_t)B_ * T_ * H_;     // 1,048,576 elements
    _Float16* Qh = (_Float16*)d_ws;               // 2 MB
    _Float16* Kh = Qh + nQKV;                     // 2 MB
    _Float16* Vh = Kh + nQKV;                     // 2 MB  (6 MB total ws use)

    qkv_proj_wmma<<<dim3((B_ * T_) / 64), dim3(128), 0, stream>>>(
        x, Wq, Wk, Wv, Qh, Kh, Vh);
    flash_attn_wmma<<<dim3(T_ / 64, B_), dim3(128), 0, stream>>>(
        Qh, Kh, Vh, out);
}
